// Attention_52166672777561
// MI455X (gfx1250) — compile-verified, matrix operand order checked by reading, not run on hardware
//
#include <hip/hip_runtime.h>
#include <hip/hip_bf16.h>

typedef __bf16 bf16_t;
typedef __attribute__((ext_vector_type(16))) __bf16 v16bf;
typedef __attribute__((ext_vector_type(8)))  float  v8f;

#define S_TOT 2304
#define S_TXT 256
#define S_IMG 2048
#define DMODEL 3072
#define HEADS 24
#define DHD 128

union Frag { v16bf v; unsigned u[8]; };

static __device__ __forceinline__ v8f wmma_bf16(const Frag& a, const Frag& b, v8f c) {
  // D = A(16x32 bf16) * B(32x16 bf16) + C(16x16 f32)
  return __builtin_amdgcn_wmma_f32_16x16x32_bf16(false, a.v, false, b.v, (short)0, c, false, false);
}

// contraction-pair index inside a 16-bit A/B fragment dword v for this half-wave
static __device__ __forceinline__ int frag_pair(int v, int half) {
  return (v & 3) + ((v >> 2) << 3) + (half << 2);
}

// ---------------------------------------------------------------------------
// K1: concat [txt; img] and convert f32 -> bf16
// ---------------------------------------------------------------------------
__global__ __launch_bounds__(256) void cvt_concat_kernel(const float* __restrict__ hid,
                                                         const float* __restrict__ enc,
                                                         bf16_t* __restrict__ X) {
  int idx = blockIdx.x * 256 + threadIdx.x;  // grid covers S_TOT*DMODEL exactly
  float v = (idx < S_TXT * DMODEL) ? enc[idx] : hid[idx - S_TXT * DMODEL];
  X[idx] = (bf16_t)v;
}

// ---------------------------------------------------------------------------
// K2: fused QKV projection GEMM (bf16 WMMA, f32 accum) + RMSNorm + RoPE
// grid: (18 row-tiles, 24 heads, 3 matrices). block 256 = 8 waves (4x2).
// Tile 128x128, K-step 64. The cold W (f32) stream for step t+1 is loaded
// into registers while step t's WMMAs run; the hot A (bf16, L2-resident)
// tile is copied directly at the top of each step.
// ---------------------------------------------------------------------------
__global__ __launch_bounds__(256) void proj_kernel(
    const bf16_t* __restrict__ X,
    const float* __restrict__ Wq, const float* __restrict__ Wk, const float* __restrict__ Wv,
    const float* __restrict__ Waq, const float* __restrict__ Wak, const float* __restrict__ Wav,
    const float* __restrict__ gq, const float* __restrict__ gk,
    const float* __restrict__ gaq, const float* __restrict__ gak,
    const float* __restrict__ freqs,   // [2304][64][2][2] f32
    bf16_t* __restrict__ qbuf,         // [24][2304][128]
    bf16_t* __restrict__ kbuf,         // [24][2304][128]
    bf16_t* __restrict__ vT)           // [24][128][2304]
{
  const int tm = blockIdx.x, head = blockIdx.y, mat = blockIdx.z;
  const bool txt = (tm < 2);  // rows 0..255 are the text stream
  const float* W = (mat == 0) ? (txt ? Waq : Wq)
                 : (mat == 1) ? (txt ? Wak : Wk)
                              : (txt ? Wav : Wv);
  const float* g = (mat == 0) ? (txt ? gaq : gq) : (txt ? gak : gk);

  __shared__ bf16_t Asm[128 * 64];     // [m][k]  bf16
  __shared__ bf16_t Bsm[128 * 64];     // [n][k]  bf16 (transposed W tile)
  __shared__ float  rowsq[2][128];
  __shared__ float  rstd[128];

  const int tid = threadIdx.x;
  const int wave = tid >> 5, lane = tid & 31, m = lane & 15, half = lane >> 4;
  const int wm = wave >> 1, wn = wave & 1;   // wave grid 4x2: 32 rows x 64 cols

  v8f acc[2][4];
  #pragma unroll
  for (int a = 0; a < 2; ++a)
    #pragma unroll
    for (int b = 0; b < 4; ++b)
      #pragma unroll
      for (int j = 0; j < 8; ++j) acc[a][b][j] = 0.0f;

  const uint4*  X4 = (const uint4*)X;        // bf16: row stride 384 uint4
  const float4* W4 = (const float4*)W;       // f32: row stride 768 float4
  uint4* Au4 = (uint4*)Asm;
  const unsigned* Au = (const unsigned*)Asm;

  // register staging for the W (cold) stream only: 8 float4 = 32 VGPRs
  float4 bReg[8];

  auto load_b = [&](int kt) {
    #pragma unroll
    for (int i = 0; i < 8; ++i) {
      int c = tid + i * 256;              // 2048 float4 = 64 rows x 32
      int k = c >> 5, n4 = c & 31;
      bReg[i] = W4[(size_t)(kt + k) * 768 + head * 32 + n4];
    }
  };
  auto store_tile = [&](int kt) {
    // A tile: 4 independent b128 loads (temps) then 4 LDS stores
    uint4 t[4];
    #pragma unroll
    for (int i = 0; i < 4; ++i) {
      int idx = tid + i * 256;
      int r = idx >> 3, c4 = idx & 7;
      t[i] = X4[(size_t)(tm * 128 + r) * 384 + (kt >> 3) + c4];
    }
    #pragma unroll
    for (int i = 0; i < 4; ++i) {
      int idx = tid + i * 256;
      int r = idx >> 3, c4 = idx & 7;
      Au4[r * 8 + c4] = t[i];
    }
    // B tile: convert pipelined registers and scatter into [n][k]
    #pragma unroll
    for (int i = 0; i < 8; ++i) {
      int c = tid + i * 256;
      int k = c >> 5, n = (c & 31) * 4;
      Bsm[(n + 0) * 64 + k] = (bf16_t)bReg[i].x;
      Bsm[(n + 1) * 64 + k] = (bf16_t)bReg[i].y;
      Bsm[(n + 2) * 64 + k] = (bf16_t)bReg[i].z;
      Bsm[(n + 3) * 64 + k] = (bf16_t)bReg[i].w;
    }
  };

  load_b(0);
  for (int kt = 0; kt < DMODEL; kt += 64) {
    store_tile(kt);
    // prefetch two K-steps ahead (global_prefetch_b8) to warm GL2
    if (kt + 128 < DMODEL) {
      int r = tid >> 2, c = (tid & 3) * 32;
      __builtin_prefetch(&W[(size_t)(kt + 128 + r) * DMODEL + head * 128 + c], 0, 3);
    }
    __syncthreads();
    if (kt + 64 < DMODEL) load_b(kt + 64);   // issue early; overlaps WMMAs below

    const unsigned* Bu = (const unsigned*)Bsm;
    #pragma unroll
    for (int ks = 0; ks < 2; ++ks) {
      Frag a[2], b[4];
      #pragma unroll
      for (int sm = 0; sm < 2; ++sm)
        #pragma unroll
        for (int v = 0; v < 8; ++v)
          a[sm].u[v] = Au[(wm * 32 + sm * 16 + m) * 32 + ks * 16 + frag_pair(v, half)];
      #pragma unroll
      for (int sn = 0; sn < 4; ++sn)
        #pragma unroll
        for (int v = 0; v < 8; ++v)
          b[sn].u[v] = Bu[(wn * 64 + sn * 16 + m) * 32 + ks * 16 + frag_pair(v, half)];
      #pragma unroll
      for (int sm = 0; sm < 2; ++sm)
        #pragma unroll
        for (int sn = 0; sn < 4; ++sn)
          acc[sm][sn] = wmma_bf16(a[sm], b[sn], acc[sm][sn]);
    }
    __syncthreads();
  }

  if (mat < 2) {
    // --- RMSNorm: per-row sum of squares (this wave covers 64 of 128 cols) ---
    #pragma unroll
    for (int sm = 0; sm < 2; ++sm) {
      float t[8];
      #pragma unroll
      for (int j = 0; j < 8; ++j) {
        float s = 0.0f;
        #pragma unroll
        for (int sn = 0; sn < 4; ++sn) { float x = acc[sm][sn][j]; s += x * x; }
        #pragma unroll
        for (int off = 1; off < 16; off <<= 1) s += __shfl_xor(s, off, 32);
        t[j] = s;
      }
      if (m == 0) {
        #pragma unroll
        for (int j = 0; j < 8; ++j)
          rowsq[wn][wm * 32 + sm * 16 + half * 8 + j] = t[j];
      }
    }
    __syncthreads();
    if (tid < 128)
      rstd[tid] = rsqrtf((rowsq[0][tid] + rowsq[1][tid]) * (1.0f / 128.0f) + 1e-5f);
    __syncthreads();

    bf16_t* dst = (mat == 0) ? qbuf : kbuf;
    #pragma unroll
    for (int sm = 0; sm < 2; ++sm)
      #pragma unroll
      for (int sn = 0; sn < 4; ++sn)
        #pragma unroll
        for (int j = 0; j < 8; ++j) {
          int row = wm * 32 + sm * 16 + half * 8 + j;   // C layout: M = j + 8*half
          int col = wn * 64 + sn * 16 + m;              // dh index, N = lane&15
          int s   = tm * 128 + row;                     // global sequence pos
          float x  = acc[sm][sn][j] * rstd[row] * g[col];
          float xp = __shfl_xor(x, 1, 32);              // dh partner (col^1)
          const float* f = freqs + ((size_t)s * 64 + (col >> 1)) * 4;
          float o = (col & 1) ? (f[2] * xp + f[3] * x)  // out[2i+1]=F10*x0+F11*x1
                              : (f[0] * x + f[1] * xp); // out[2i]  =F00*x0+F01*x1
          dst[((size_t)head * S_TOT + s) * DHD + col] = (bf16_t)o;
        }
  } else {
    // --- V: store transposed [head][dh][s] for the PV B-fragments ---
    #pragma unroll
    for (int sm = 0; sm < 2; ++sm)
      #pragma unroll
      for (int sn = 0; sn < 4; ++sn)
        #pragma unroll
        for (int j = 0; j < 8; ++j) {
          int row = wm * 32 + sm * 16 + half * 8 + j;
          int col = wn * 64 + sn * 16 + m;
          vT[((size_t)head * DHD + col) * S_TOT + tm * 128 + row] = (bf16_t)acc[sm][sn][j];
        }
  }
}

// ---------------------------------------------------------------------------
// K3: flash attention, one (head, 128-row q-tile) per workgroup.
// Each wave owns 16 q rows; key loop in chunks of 32; WMMA for QK^T and PV.
// ---------------------------------------------------------------------------
__global__ __launch_bounds__(256) void attn_kernel(
    const bf16_t* __restrict__ qbuf, const bf16_t* __restrict__ kbuf,
    const bf16_t* __restrict__ vT, bf16_t* __restrict__ attn)  // attn [2304][3072]
{
  const int qt = blockIdx.x, head = blockIdx.y;
  const int tid = threadIdx.x, wave = tid >> 5, lane = tid & 31;
  const int m = lane & 15, half = lane >> 4;
  const int qrow0 = qt * 128 + wave * 16;

  __shared__ bf16_t pstage[8][16 * 32];   // per-wave P staging (C-layout -> A-layout)

  const unsigned* qd = (const unsigned*)qbuf + (size_t)head * S_TOT * 64;
  const unsigned* kd = (const unsigned*)kbuf + (size_t)head * S_TOT * 64;
  const unsigned* vd = (const unsigned*)vT   + (size_t)head * DHD * (S_TOT / 2);
  unsigned* pu = (unsigned*)&pstage[wave][0];

  // Q A-fragments: 16 rows x 128 dh = 4 fragments of 16x32
  Frag qa[4];
  #pragma unroll
  for (int dc = 0; dc < 4; ++dc)
    #pragma unroll
    for (int v = 0; v < 8; ++v)
      qa[dc].u[v] = qd[(qrow0 + m) * 64 + dc * 16 + frag_pair(v, half)];

  v8f mrow, lrow, O[8];
  #pragma unroll
  for (int j = 0; j < 8; ++j) { mrow[j] = -1.0e30f; lrow[j] = 0.0f; }
  #pragma unroll
  for (int nb = 0; nb < 8; ++nb)
    #pragma unroll
    for (int j = 0; j < 8; ++j) O[nb][j] = 0.0f;

  const float scale = 0.08838834764831845f;  // 1/sqrt(128)

  for (int kc = 0; kc < S_TOT; kc += 32) {
    // ---- scores: two 16x16 C tiles (keys kc..kc+15, kc+16..kc+31) ----
    v8f c0, c1;
    #pragma unroll
    for (int j = 0; j < 8; ++j) { c0[j] = 0.0f; c1[j] = 0.0f; }
    #pragma unroll
    for (int dc = 0; dc < 4; ++dc) {
      Frag b0, b1;
      #pragma unroll
      for (int v = 0; v < 8; ++v) {
        int pr = frag_pair(v, half);
        b0.u[v] = kd[(kc + m) * 64 + dc * 16 + pr];        // B[d][n] = K[key n][d]
        b1.u[v] = kd[(kc + 16 + m) * 64 + dc * 16 + pr];
      }
      c0 = wmma_bf16(qa[dc], b0, c0);
      c1 = wmma_bf16(qa[dc], b1, c1);
    }

    // ---- online softmax (row = j + 8*half, cols across lanes in half-wave) ----
    float alpha[8], p0[8], p1[8];
    #pragma unroll
    for (int j = 0; j < 8; ++j) {
      float a = c0[j] * scale, b = c1[j] * scale;
      float t = fmaxf(a, b);
      #pragma unroll
      for (int off = 1; off < 16; off <<= 1) t = fmaxf(t, __shfl_xor(t, off, 32));
      float mn = fmaxf(mrow[j], t);
      alpha[j] = __expf(mrow[j] - mn);
      p0[j] = __expf(a - mn);
      p1[j] = __expf(b - mn);
      float r = p0[j] + p1[j];
      #pragma unroll
      for (int off = 1; off < 16; off <<= 1) r += __shfl_xor(r, off, 32);
      lrow[j] = lrow[j] * alpha[j] + r;
      mrow[j] = mn;
    }
    #pragma unroll
    for (int nb = 0; nb < 8; ++nb)
      #pragma unroll
      for (int j = 0; j < 8; ++j) O[nb][j] *= alpha[j];

    // ---- P: C-layout -> A-layout via per-wave LDS tile (intra-wave, in-order) ----
    #pragma unroll
    for (int j = 0; j < 8; ++j) {
      pstage[wave][(half * 8 + j) * 32 + m]      = (bf16_t)p0[j];
      pstage[wave][(half * 8 + j) * 32 + m + 16] = (bf16_t)p1[j];
    }
    asm volatile("" ::: "memory");
    Frag pa;
    #pragma unroll
    for (int v = 0; v < 8; ++v)
      pa.u[v] = pu[m * 16 + frag_pair(v, half)];
    asm volatile("" ::: "memory");

    // ---- O += P(16x32) * V(32x128): 8 WMMAs over dh column blocks ----
    #pragma unroll
    for (int nb = 0; nb < 8; ++nb) {
      Frag vb;
      #pragma unroll
      for (int v = 0; v < 8; ++v)
        vb.u[v] = vd[(nb * 16 + m) * 1152 + (kc >> 1) + frag_pair(v, half)];
      O[nb] = wmma_bf16(pa, vb, O[nb]);
    }
  }

  // ---- finalize: divide by l, write bf16 into attn [s][head*128 + dh] ----
  #pragma unroll
  for (int nb = 0; nb < 8; ++nb)
    #pragma unroll
    for (int j = 0; j < 8; ++j) {
      int row = half * 8 + j;
      int s = qrow0 + row;
      float o = O[nb][j] / lrow[j];
      attn[(size_t)s * DMODEL + head * DHD + nb * 16 + m] = (bf16_t)o;
    }
}

// ---------------------------------------------------------------------------
// K4: output projections (per-stream weight/bias), f32 output.
// grid: (18 row-tiles, 24 col-tiles of 128). Same pipeline as K2.
// ---------------------------------------------------------------------------
__global__ __launch_bounds__(256) void oproj_kernel(
    const bf16_t* __restrict__ A,      // attn [2304][3072] bf16
    const float* __restrict__ Wo, const float* __restrict__ bo,
    const float* __restrict__ Wao, const float* __restrict__ bao,
    float* __restrict__ out)           // img [2048][3072] then txt [256][3072]
{
  const int tm = blockIdx.x, tn = blockIdx.y;
  const bool txt = (tm < 2);
  const float* W    = txt ? Wao : Wo;
  const float* bias = txt ? bao : bo;

  __shared__ bf16_t Asm[128 * 64];
  __shared__ bf16_t Bsm[128 * 64];

  const int tid = threadIdx.x;
  const int wave = tid >> 5, lane = tid & 31, m = lane & 15, half = lane >> 4;
  const int wm = wave >> 1, wn = wave & 1;

  v8f acc[2][4];
  #pragma unroll
  for (int a = 0; a < 2; ++a)
    #pragma unroll
    for (int b = 0; b < 4; ++b)
      #pragma unroll
      for (int j = 0; j < 8; ++j) acc[a][b][j] = 0.0f;

  const uint4*  A4 = (const uint4*)A;
  const float4* W4 = (const float4*)W;
  uint4* Au4 = (uint4*)Asm;
  const unsigned* Au = (const unsigned*)Asm;

  float4 bReg[8];

  auto load_b = [&](int kt) {
    #pragma unroll
    for (int i = 0; i < 8; ++i) {
      int c = tid + i * 256;
      int k = c >> 5, n4 = c & 31;
      bReg[i] = W4[(size_t)(kt + k) * 768 + tn * 32 + n4];
    }
  };
  auto store_tile = [&](int kt) {
    uint4 t[4];
    #pragma unroll
    for (int i = 0; i < 4; ++i) {
      int idx = tid + i * 256;
      int r = idx >> 3, c4 = idx & 7;
      t[i] = A4[(size_t)(tm * 128 + r) * 384 + (kt >> 3) + c4];
    }
    #pragma unroll
    for (int i = 0; i < 4; ++i) {
      int idx = tid + i * 256;
      int r = idx >> 3, c4 = idx & 7;
      Au4[r * 8 + c4] = t[i];
    }
    #pragma unroll
    for (int i = 0; i < 8; ++i) {
      int c = tid + i * 256;
      int k = c >> 5, n = (c & 31) * 4;
      Bsm[(n + 0) * 64 + k] = (bf16_t)bReg[i].x;
      Bsm[(n + 1) * 64 + k] = (bf16_t)bReg[i].y;
      Bsm[(n + 2) * 64 + k] = (bf16_t)bReg[i].z;
      Bsm[(n + 3) * 64 + k] = (bf16_t)bReg[i].w;
    }
  };

  load_b(0);
  for (int kt = 0; kt < DMODEL; kt += 64) {
    store_tile(kt);
    if (kt + 128 < DMODEL) {
      int r = tid >> 2, c = (tid & 3) * 32;
      __builtin_prefetch(&W[(size_t)(kt + 128 + r) * DMODEL + tn * 128 + c], 0, 3);
    }
    __syncthreads();
    if (kt + 64 < DMODEL) load_b(kt + 64);

    const unsigned* Bu = (const unsigned*)Bsm;
    #pragma unroll
    for (int ks = 0; ks < 2; ++ks) {
      Frag a[2], b[4];
      #pragma unroll
      for (int sm = 0; sm < 2; ++sm)
        #pragma unroll
        for (int v = 0; v < 8; ++v)
          a[sm].u[v] = Au[(wm * 32 + sm * 16 + m) * 32 + ks * 16 + frag_pair(v, half)];
      #pragma unroll
      for (int sn = 0; sn < 4; ++sn)
        #pragma unroll
        for (int v = 0; v < 8; ++v)
          b[sn].u[v] = Bu[(wn * 64 + sn * 16 + m) * 32 + ks * 16 + frag_pair(v, half)];
      #pragma unroll
      for (int sm = 0; sm < 2; ++sm)
        #pragma unroll
        for (int sn = 0; sn < 4; ++sn)
          acc[sm][sn] = wmma_bf16(a[sm], b[sn], acc[sm][sn]);
    }
    __syncthreads();
  }

  #pragma unroll
  for (int sm = 0; sm < 2; ++sm)
    #pragma unroll
    for (int sn = 0; sn < 4; ++sn)
      #pragma unroll
      for (int j = 0; j < 8; ++j) {
        int row  = wm * 32 + sm * 16 + half * 8 + j;
        int colg = tn * 128 + wn * 64 + sn * 16 + m;
        int rowg = tm * 128 + row;
        float v = acc[sm][sn][j] + bias[colg];
        if (txt)
          out[(size_t)S_IMG * DMODEL + (size_t)rowg * DMODEL + colg] = v;
        else
          out[(size_t)(rowg - S_TXT) * DMODEL + colg] = v;
      }
}

// ---------------------------------------------------------------------------
extern "C" void kernel_launch(void* const* d_in, const int* in_sizes, int n_in,
                              void* d_out, int out_size, void* d_ws, size_t ws_size,
                              hipStream_t stream) {
  (void)in_sizes; (void)n_in; (void)out_size; (void)ws_size;

  const float* hid  = (const float*)d_in[0];
  const float* enc  = (const float*)d_in[1];
  const float* rope = (const float*)d_in[2];
  const float* Wq   = (const float*)d_in[3];
  const float* Wk   = (const float*)d_in[4];
  const float* Wv   = (const float*)d_in[5];
  const float* Waq  = (const float*)d_in[6];
  const float* Wak  = (const float*)d_in[7];
  const float* Wav  = (const float*)d_in[8];
  const float* gq   = (const float*)d_in[9];
  const float* gk   = (const float*)d_in[10];
  const float* gaq  = (const float*)d_in[11];
  const float* gak  = (const float*)d_in[12];
  const float* Wo   = (const float*)d_in[13];
  const float* bo   = (const float*)d_in[14];
  const float* Wao  = (const float*)d_in[15];
  const float* bao  = (const float*)d_in[16];
  float* out = (float*)d_out;

  char* ws = (char*)d_ws;
  size_t off = 0;
  bf16_t* X    = (bf16_t*)(ws + off); off += (size_t)S_TOT * DMODEL * 2;
  bf16_t* qbuf = (bf16_t*)(ws + off); off += (size_t)HEADS * S_TOT * DHD * 2;
  bf16_t* kbuf = (bf16_t*)(ws + off); off += (size_t)HEADS * S_TOT * DHD * 2;
  bf16_t* vT   = (bf16_t*)(ws + off); off += (size_t)HEADS * S_TOT * DHD * 2;
  bf16_t* attn = (bf16_t*)(ws + off); off += (size_t)S_TOT * DMODEL * 2;

  cvt_concat_kernel<<<(S_TOT * DMODEL) / 256, 256, 0, stream>>>(hid, enc, X);
  proj_kernel<<<dim3(18, 24, 3), 256, 0, stream>>>(X, Wq, Wk, Wv, Waq, Wak, Wav,
                                                   gq, gk, gaq, gak, rope,
                                                   qbuf, kbuf, vT);
  attn_kernel<<<dim3(18, 24), 256, 0, stream>>>(qbuf, kbuf, vT, attn);
  oproj_kernel<<<dim3(18, 24), 256, 0, stream>>>(attn, Wo, bo, Wao, bao, out);
}
